// LightweightTransformerLayer_81827716923838
// MI455X (gfx1250) — matrix operand order repaired and checked by reading, compiled, not run on hardware
//
#include <hip/hip_runtime.h>
#include <hip/hip_bf16.h>
#include <math.h>

// ---------------------------------------------------------------------------
// LightweightTransformerLayer fused as 4 kernels for gfx1250 (MI455X).
// All GEMMs use v_wmma_f32_16x16x32_bf16 (wave32, f32 accumulate).
// Weights pre-converted to bf16 once (k_cvt) so B-fragments are contiguous
// bf16 b128 loads; V stored transposed in LDS so probs@V B-fragments merge
// into ds_load_b128; async global->LDS copies (gfx1250 ASYNCcnt path).
// ---------------------------------------------------------------------------

typedef __bf16 bf16;
typedef __attribute__((ext_vector_type(16))) __bf16 v16bf;
typedef __attribute__((ext_vector_type(8)))  float  v8f;

#define DIMC   256
#define NHEAD  4
#define HDIM   64
#define HIDDEN 512
#define PTOK   256
#define NSEQ   64
#define ATT_SCALE 0.125f   // 64^-0.5
#define LN_EPS 1e-5f

#define LDH  264   // C=256 + 8 pad
#define LDQ  72    // 64 + 8 pad
#define LDS_SC 264
#define LDVT 264   // vT cols (256 keys + 8)
#define LDM  520   // 512 + 8 pad

// ---------------- optional gfx1250 async global->LDS path ------------------
#if __has_builtin(__builtin_amdgcn_global_load_async_to_lds_b128) && \
    __has_builtin(__builtin_amdgcn_global_load_async_to_lds_b32)  && \
    __has_builtin(__builtin_amdgcn_s_wait_asynccnt)
#define HAVE_ASYNC_LDS 1
// exact pointee types per compiler diagnostics:
//   b128: 'int __attribute__((vector_size(16))) __device__ *'
//   b32 : 'int __device__ *'
typedef int v4i_vs __attribute__((vector_size(16)));
typedef __attribute__((address_space(1))) v4i_vs* g_b128_t;
typedef __attribute__((address_space(3))) v4i_vs* l_b128_t;
typedef __attribute__((address_space(1))) int*    g_b32_t;
typedef __attribute__((address_space(3))) int*    l_b32_t;
// generic LDS address: low 32 bits are the LDS byte offset (ISA 10.2)
__device__ __forceinline__ g_b128_t gb128(const void* p) {
  return (g_b128_t)(__UINTPTR_TYPE__)p;
}
__device__ __forceinline__ l_b128_t lb128(const void* p) {
  return (l_b128_t)(__UINT32_TYPE__)(__UINTPTR_TYPE__)p;
}
__device__ __forceinline__ g_b32_t gb32(const void* p) {
  return (g_b32_t)(__UINTPTR_TYPE__)p;
}
__device__ __forceinline__ l_b32_t lb32(const void* p) {
  return (l_b32_t)(__UINT32_TYPE__)(__UINTPTR_TYPE__)p;
}
#else
#define HAVE_ASYNC_LDS 0
#endif

// ---------------- WMMA fragment helpers (A/B: contiguous k = 16*(lane/16)+e; D: ISA 7.12.2 layout) ----------------

// A: 16x32 bf16, row-major, ld elements/row.
__device__ __forceinline__ v16bf frag_A_lds(const bf16* __restrict__ base, int ld) {
  const int lane = threadIdx.x & 31;
  const int m  = lane & 15;
  const int kb = (lane >> 4) << 4;
  const bf16* row = base + m * ld;
  v16bf a;
#pragma unroll
  for (int e = 0; e < 16; ++e) {
    const int k = kb + e;
    a[e] = row[k];
  }
  return a;
}

// B: logical B[k][n] where source stored N x K (contiguous K per n-row).
__device__ __forceinline__ v16bf frag_B_nk(const bf16* __restrict__ base, int ld) {
  const int lane = threadIdx.x & 31;
  const int n  = lane & 15;
  const int kb = (lane >> 4) << 4;
  const bf16* row = base + n * ld + kb;   // 16 contiguous bf16 -> 2x b128
  v16bf b;
#pragma unroll
  for (int e = 0; e < 16; ++e)
    b[e] = row[e];
  return b;
}

// D (f32 16x16): n = lane%16, m = r + 8*(lane>=16)
__device__ __forceinline__ void store_D_bf16(bf16* __restrict__ base, int ld, v8f c) {
  const int lane = threadIdx.x & 31;
  const int n  = lane & 15;
  const int mb = (lane >> 4) << 3;
#pragma unroll
  for (int r = 0; r < 8; ++r)
    base[(mb + r) * ld + n] = (bf16)c[r];
}

// D stored transposed: D[m][n] -> base[n*ld + m]
__device__ __forceinline__ void store_D_bf16_T(bf16* __restrict__ base, int ld, v8f c) {
  const int lane = threadIdx.x & 31;
  const int n  = lane & 15;
  const int mb = (lane >> 4) << 3;
#pragma unroll
  for (int r = 0; r < 8; ++r)
    base[n * ld + mb + r] = (bf16)c[r];
}

__device__ __forceinline__ void store_D_f32(float* __restrict__ base, int ld, v8f c, float s) {
  const int lane = threadIdx.x & 31;
  const int n  = lane & 15;
  const int mb = (lane >> 4) << 3;
#pragma unroll
  for (int r = 0; r < 8; ++r)
    base[(mb + r) * ld + n] = c[r] * s;
}

#define WMMA_BF16(A, B, C) \
  __builtin_amdgcn_wmma_f32_16x16x32_bf16(false, (A), false, (B), (short)0, (C), false, false)

// ---------------------------------------------------------------------------
// Kernel 0: convert all weights f32 -> bf16 (same [out,in] layout) into ws.
// ---------------------------------------------------------------------------
__global__ __launch_bounds__(256) void k_cvt(
    const float* __restrict__ Wq, const float* __restrict__ Wp,
    const float* __restrict__ W1, const float* __restrict__ W2,
    bf16* __restrict__ wq, bf16* __restrict__ wp,
    bf16* __restrict__ w1, bf16* __restrict__ w2) {
  const int i = blockIdx.x * 256 + threadIdx.x;
  const int stride = gridDim.x * 256;
  for (int j = i; j < 768 * DIMC;    j += stride) wq[j] = (bf16)Wq[j];
  for (int j = i; j < DIMC * DIMC;   j += stride) wp[j] = (bf16)Wp[j];
  for (int j = i; j < HIDDEN * DIMC; j += stride) w1[j] = (bf16)W1[j];
  for (int j = i; j < DIMC * HIDDEN; j += stride) w2[j] = (bf16)W2[j];
}

// ---------------------------------------------------------------------------
// Kernel 1: LN1 + per-head QKV + softmax(QK^T)V -> o_ws[seq][p][c] bf16
// grid (512, 4 heads), 256 threads. LDS ~287 KB.
// ---------------------------------------------------------------------------
__global__ __launch_bounds__(256) void k_attn(
    const float* __restrict__ x, const bf16* __restrict__ wqkv,
    const float* __restrict__ ln1_g, const float* __restrict__ ln1_b,
    bf16* __restrict__ o_ws) {
  extern __shared__ char smem[];
  float* scores = (float*)smem;                              // 32 x 264 f32
  bf16*  hbuf   = (bf16*)(smem + 32 * LDS_SC * 4);           // 256 x 264
  bf16*  qh     = hbuf + PTOK * LDH;                         // 256 x 72
  bf16*  kh     = qh   + PTOK * LDQ;                         // 256 x 72
  bf16*  vT     = kh   + PTOK * LDQ;                         // 64  x 264 (transposed V)
  bf16*  probs  = vT   + HDIM * LDVT;                        // 32  x 264

  const int seq  = blockIdx.x;
  const int head = blockIdx.y;
  const int bb   = seq >> 6;
  const int nn   = seq & 63;
  const int tid  = threadIdx.x;
  const int wv   = tid >> 5;

  // ---- load x ([B,C,N,P] gather, coalesced along P) as bf16 ----
  for (int idx = tid; idx < PTOK * DIMC; idx += 256) {
    const int c = idx >> 8, p = idx & 255;
    const size_t g = (((size_t)bb * DIMC + c) * NSEQ + nn) * PTOK + p;
    hbuf[p * LDH + c] = (bf16)x[g];
  }
  __syncthreads();

  // ---- LN1, one thread per token row ----
  {
    bf16* row = hbuf + tid * LDH;
    float mu = 0.f;
#pragma unroll 8
    for (int c = 0; c < DIMC; ++c) mu += (float)row[c];
    mu *= (1.0f / DIMC);
    float var = 0.f;
#pragma unroll 8
    for (int c = 0; c < DIMC; ++c) { float d = (float)row[c] - mu; var += d * d; }
    var *= (1.0f / DIMC);
    const float rs = rsqrtf(var + LN_EPS);
#pragma unroll 8
    for (int c = 0; c < DIMC; ++c)
      row[c] = (bf16)(((float)row[c] - mu) * rs * ln1_g[c] + ln1_b[c]);
  }
  __syncthreads();

  // ---- QKV GEMMs, jobs: mat(3) x mtile(16) x ntile(4) = 192 ----
  for (int job = wv; job < 192; job += 8) {
    const int mat = job >> 6;
    const int mt  = (job & 63) >> 2;
    const int nt  = job & 3;
    const bf16* wrow = wqkv + (size_t)(mat * DIMC + head * HDIM + nt * 16) * DIMC;
    if (job + 8 < 192) {
      const int j2 = job + 8;
      __builtin_prefetch(wqkv + (size_t)((j2 >> 6) * DIMC + head * HDIM + (j2 & 3) * 16) * DIMC, 0, 1);
    }
    v8f acc = 0.0f;
#pragma unroll
    for (int kt = 0; kt < 8; ++kt) {
      v16bf a = frag_A_lds(hbuf + (mt * 16) * LDH + kt * 32, LDH);
      v16bf b = frag_B_nk(wrow + kt * 32, DIMC);
      acc = WMMA_BF16(a, b, acc);
    }
    if (mat == 2)      store_D_bf16_T(vT + (nt * 16) * LDVT + mt * 16, LDVT, acc);
    else if (mat == 1) store_D_bf16(kh + (mt * 16) * LDQ + nt * 16, LDQ, acc);
    else               store_D_bf16(qh + (mt * 16) * LDQ + nt * 16, LDQ, acc);
  }
  __syncthreads();

  // ---- attention, 32-query blocks ----
  for (int qb = 0; qb < 8; ++qb) {
    // scores = Q_blk K^T * scale ; B[d][key] = K[key][d] (contiguous in kh)
    for (int job = wv; job < 32; job += 8) {
      const int mt = job >> 4, nt = job & 15;
      v8f acc = 0.0f;
#pragma unroll
      for (int kt = 0; kt < 2; ++kt) {
        v16bf a = frag_A_lds(qh + (qb * 32 + mt * 16) * LDQ + kt * 32, LDQ);
        v16bf b = frag_B_nk(kh + (nt * 16) * LDQ + kt * 32, LDQ);
        acc = WMMA_BF16(a, b, acc);
      }
      store_D_f32(scores + (mt * 16) * LDS_SC + nt * 16, LDS_SC, acc, ATT_SCALE);
    }
    __syncthreads();

    // softmax: 8 lanes per row, shfl_xor butterflies (width 8, in-wave)
    {
      const int row = tid >> 3;      // 0..31
      const int sub = tid & 7;
      float* srow = scores + row * LDS_SC;
      float mx = -3.4e38f;
      for (int j = sub; j < PTOK; j += 8) mx = fmaxf(mx, srow[j]);
#pragma unroll
      for (int d = 1; d < 8; d <<= 1) mx = fmaxf(mx, __shfl_xor(mx, d, 8));
      float sum = 0.f;
      for (int j = sub; j < PTOK; j += 8) {
        float e = __expf(srow[j] - mx);
        srow[j] = e; sum += e;
      }
#pragma unroll
      for (int d = 1; d < 8; d <<= 1) sum += __shfl_xor(sum, d, 8);
      const float inv = 1.0f / sum;
      for (int j = sub; j < PTOK; j += 8)
        probs[row * LDS_SC + j] = (bf16)(srow[j] * inv);
    }
    __syncthreads();

    // O_blk = probs @ V ; B[key][d] = vT[d][key] (contiguous in vT)
    {
      const int mt = wv >> 2, nt = wv & 3;
      v8f acc = 0.0f;
#pragma unroll
      for (int kt = 0; kt < 8; ++kt) {
        v16bf a = frag_A_lds(probs + (mt * 16) * LDS_SC + kt * 32, LDS_SC);
        v16bf b = frag_B_nk(vT + (nt * 16) * LDVT + kt * 32, LDVT);
        acc = WMMA_BF16(a, b, acc);
      }
      const int lane = tid & 31;
      const int n  = lane & 15;
      const int mb = (lane >> 4) << 3;
#pragma unroll
      for (int r = 0; r < 8; ++r) {
        const int p  = qb * 32 + mt * 16 + mb + r;
        const int ch = head * HDIM + nt * 16 + n;
        o_ws[((size_t)seq * PTOK + p) * DIMC + ch] = (bf16)acc[r];
      }
    }
    __syncthreads();
  }
}

// ---------------------------------------------------------------------------
// Kernel 2: x2 = x + O @ W_proj^T + b_proj -> d_out ([B,C,N,P])
// ---------------------------------------------------------------------------
__global__ __launch_bounds__(256) void k_proj(
    const float* __restrict__ x, const bf16* __restrict__ wproj,
    const float* __restrict__ b_proj, const bf16* __restrict__ o_ws,
    float* __restrict__ out) {
  extern __shared__ char smem[];
  bf16* obuf = (bf16*)smem;                                  // 256 x 264

  const int seq = blockIdx.x;
  const int bb = seq >> 6, nn = seq & 63;
  const int tid = threadIdx.x, wv = tid >> 5;

#if HAVE_ASYNC_LDS
#pragma unroll 4
  for (int i = 0; i < 32; ++i) {                 // 8192 x 16B chunks
    const int chunk = i * 256 + tid;
    const int p = chunk >> 5, cb = (chunk & 31) << 3;
    const bf16* gsrc = o_ws + ((size_t)seq * PTOK + p) * DIMC + cb;
    __builtin_amdgcn_global_load_async_to_lds_b128(gb128(gsrc), lb128(obuf + p * LDH + cb), 0, 0);
  }
  __builtin_amdgcn_s_wait_asynccnt(0);
#else
  for (int idx = tid; idx < PTOK * DIMC; idx += 256) {
    const int p = idx >> 8, c = idx & 255;
    obuf[p * LDH + c] = o_ws[((size_t)seq * PTOK + p) * DIMC + c];
  }
#endif
  __syncthreads();

  for (int job = wv; job < 256; job += 8) {
    const int mt = job >> 4, nt = job & 15;
    if (job + 8 < 256) __builtin_prefetch(wproj + (size_t)(((job + 8) & 15) * 16) * DIMC, 0, 1);
    v8f acc = 0.0f;
#pragma unroll
    for (int kt = 0; kt < 8; ++kt) {
      v16bf a = frag_A_lds(obuf + (mt * 16) * LDH + kt * 32, LDH);
      v16bf b = frag_B_nk(wproj + (size_t)(nt * 16) * DIMC + kt * 32, DIMC);
      acc = WMMA_BF16(a, b, acc);
    }
    const int lane = tid & 31;
    const int n  = lane & 15;
    const int mb = (lane >> 4) << 3;
#pragma unroll
    for (int r = 0; r < 8; ++r) {
      const int p  = mt * 16 + mb + r;
      const int ch = nt * 16 + n;
      const size_t g = (((size_t)bb * DIMC + ch) * NSEQ + nn) * PTOK + p;
      out[g] = x[g] + acc[r] + b_proj[ch];
    }
  }
}

// ---------------------------------------------------------------------------
// Kernel 3: d_out += MLP(LN2(d_out)) in place, per (seq, 64-token block)
// ---------------------------------------------------------------------------
__global__ __launch_bounds__(256) void k_mlp(
    const float* __restrict__ ln2_g, const float* __restrict__ ln2_b,
    const bf16* __restrict__ w1, const float* __restrict__ b_mlp1,
    const bf16* __restrict__ w2, const float* __restrict__ b_mlp2,
    float* __restrict__ out) {
  extern __shared__ char smem[];
  float* xb  = (float*)smem;                                 // 64 x 264 f32
  bf16*  h2  = (bf16*)(smem + 64 * LDH * 4);                 // 64 x 264
  bf16*  mb_ = h2 + 64 * LDH;                                // 64 x 520

  const int seq = blockIdx.x, tb = blockIdx.y;
  const int bb = seq >> 6, nn = seq & 63;
  const int tid = threadIdx.x, wv = tid >> 5;

  // load x2 block, consecutive tids walk contiguous P
#if HAVE_ASYNC_LDS
#pragma unroll 4
  for (int i = 0; i < 64; ++i) {
    const int idx = i * 256 + tid;
    const int c = idx >> 6, r = idx & 63;
    const size_t g = (((size_t)bb * DIMC + c) * NSEQ + nn) * PTOK + tb * 64 + r;
    __builtin_amdgcn_global_load_async_to_lds_b32(gb32(out + g), lb32(xb + r * LDH + c), 0, 0);
  }
  __builtin_amdgcn_s_wait_asynccnt(0);
#else
  for (int idx = tid; idx < 64 * DIMC; idx += 256) {
    const int c = idx >> 6, r = idx & 63;
    const size_t g = (((size_t)bb * DIMC + c) * NSEQ + nn) * PTOK + tb * 64 + r;
    xb[r * LDH + c] = out[g];
  }
#endif
  __syncthreads();

  // LN2: 4 lanes per row
  {
    const int row = tid >> 2, sub = tid & 3;
    const float* xr = xb + row * LDH;
    float s = 0.f;
    for (int c = sub; c < DIMC; c += 4) s += xr[c];
#pragma unroll
    for (int d = 1; d < 4; d <<= 1) s += __shfl_xor(s, d, 4);
    const float mu = s * (1.0f / DIMC);
    float v = 0.f;
    for (int c = sub; c < DIMC; c += 4) { float t = xr[c] - mu; v += t * t; }
#pragma unroll
    for (int d = 1; d < 4; d <<= 1) v += __shfl_xor(v, d, 4);
    const float rs = rsqrtf(v * (1.0f / DIMC) + LN_EPS);
    for (int c = sub; c < DIMC; c += 4)
      h2[row * LDH + c] = (bf16)((xr[c] - mu) * rs * ln2_g[c] + ln2_b[c]);
  }
  __syncthreads();

  // GEMM1 + bias + exact GELU -> mb_ (bf16)
  for (int job = wv; job < 128; job += 8) {
    const int mt = job >> 5, nt = job & 31;
    if (job + 8 < 128) __builtin_prefetch(w1 + (size_t)(((job + 8) & 31) * 16) * DIMC, 0, 1);
    v8f acc = 0.0f;
#pragma unroll
    for (int kt = 0; kt < 8; ++kt) {
      v16bf a = frag_A_lds(h2 + (mt * 16) * LDH + kt * 32, LDH);
      v16bf b = frag_B_nk(w1 + (size_t)(nt * 16) * DIMC + kt * 32, DIMC);
      acc = WMMA_BF16(a, b, acc);
    }
    const int lane = tid & 31;
    const int n  = lane & 15;
    const int mbr = (lane >> 4) << 3;
#pragma unroll
    for (int r = 0; r < 8; ++r) {
      const int col = nt * 16 + n;
      float v = acc[r] + b_mlp1[col];
      v = 0.5f * v * (1.0f + erff(v * 0.70710678f));
      mb_[(mt * 16 + mbr + r) * LDM + col] = (bf16)v;
    }
  }
  __syncthreads();

  // GEMM2 + bias + residual -> out
  for (int job = wv; job < 64; job += 8) {
    const int mt = job >> 4, nt = job & 15;
    v8f acc = 0.0f;
#pragma unroll
    for (int kt = 0; kt < 16; ++kt) {
      v16bf a = frag_A_lds(mb_ + (mt * 16) * LDM + kt * 32, LDM);
      v16bf b = frag_B_nk(w2 + (size_t)(nt * 16) * HIDDEN + kt * 32, HIDDEN);
      acc = WMMA_BF16(a, b, acc);
    }
    const int lane = tid & 31;
    const int n  = lane & 15;
    const int mbr = (lane >> 4) << 3;
#pragma unroll
    for (int r = 0; r < 8; ++r) {
      const int row = mt * 16 + mbr + r;
      const int p  = tb * 64 + row;
      const int ch = nt * 16 + n;
      const size_t g = (((size_t)bb * DIMC + ch) * NSEQ + nn) * PTOK + p;
      out[g] = xb[row * LDH + ch] + acc[r] + b_mlp2[ch];
    }
  }
}

// ---------------------------------------------------------------------------
extern "C" void kernel_launch(void* const* d_in, const int* in_sizes, int n_in,
                              void* d_out, int out_size, void* d_ws, size_t ws_size,
                              hipStream_t stream) {
  const float* x      = (const float*)d_in[0];
  const float* W_qkv  = (const float*)d_in[1];
  const float* W_proj = (const float*)d_in[2];
  const float* b_proj = (const float*)d_in[3];
  const float* ln1_g  = (const float*)d_in[4];
  const float* ln1_b  = (const float*)d_in[5];
  const float* ln2_g  = (const float*)d_in[6];
  const float* ln2_b  = (const float*)d_in[7];
  const float* W_mlp1 = (const float*)d_in[8];
  const float* b_mlp1 = (const float*)d_in[9];
  const float* W_mlp2 = (const float*)d_in[10];
  const float* b_mlp2 = (const float*)d_in[11];
  float* out = (float*)d_out;

  // workspace layout (bf16): o[512*256*256] | wqkv | wproj | wmlp1 | wmlp2
  bf16* o_ws    = (bf16*)d_ws;
  bf16* wqkv_b  = o_ws + (size_t)512 * PTOK * DIMC;
  bf16* wproj_b = wqkv_b + (size_t)768 * DIMC;
  bf16* wmlp1_b = wproj_b + (size_t)DIMC * DIMC;
  bf16* wmlp2_b = wmlp1_b + (size_t)HIDDEN * DIMC;

  const size_t lds_attn = (size_t)32 * LDS_SC * 4        // scores
                        + (size_t)PTOK * LDH * 2         // hbuf
                        + (size_t)2 * PTOK * LDQ * 2     // q, k
                        + (size_t)HDIM * LDVT * 2        // vT
                        + (size_t)32 * LDS_SC * 2;       // probs
  const size_t lds_proj = (size_t)PTOK * LDH * 2;
  const size_t lds_mlp  = (size_t)64 * LDH * 4 + (size_t)64 * LDH * 2
                        + (size_t)64 * LDM * 2;

  k_cvt <<<dim3(256), 256, 0, stream>>>(W_qkv, W_proj, W_mlp1, W_mlp2,
                                        wqkv_b, wproj_b, wmlp1_b, wmlp2_b);
  k_attn<<<dim3(512, NHEAD), 256, lds_attn, stream>>>(x, wqkv_b, ln1_g, ln1_b, o_ws);
  k_proj<<<dim3(512), 256, lds_proj, stream>>>(x, wproj_b, b_proj, o_ws, out);
  k_mlp <<<dim3(512, 4), 256, lds_mlp, stream>>>(ln2_g, ln2_b, wmlp1_b, b_mlp1,
                                                 wmlp2_b, b_mlp2, out);
}
